// MultiheadLatentAttention_84310208021214
// MI455X (gfx1250) — matrix operand order repaired and checked by reading, not compiled, not run on hardware
//
#include <hip/hip_runtime.h>

// ---------------------------------------------------------------------------
// MLA forward for MI455X (gfx1250, wave32, WMMA).
// All matmuls in bf16 via v_wmma_f32_16x16x32_bf16, f32 accumulation.
// GEMM and flash-attention tiles staged into LDS via TDM tensor_load_to_lds
// (2-D strided descriptors) when the builtin is available.
// ---------------------------------------------------------------------------

typedef __attribute__((ext_vector_type(16))) __bf16 v16bf;
typedef __attribute__((ext_vector_type(8)))  float  v8f;
typedef unsigned int u32x4 __attribute__((ext_vector_type(4)));
typedef int          i32x8 __attribute__((ext_vector_type(8)));
typedef int          i32x4 __attribute__((ext_vector_type(4)));

#define D_MODEL 2048
#define N_HEADS 16
#define Q_LORA  1536
#define KV_LORA 512
#define NOPE    128
#define ROPE    64
#define V_DIM   128
#define BB      2
#define TT      2048
#define TOK     (BB*TT)           // 4096
#define HD      (NOPE + ROPE)     // 192

// Flip to 0 to fall back to plain vector loads if the TDM builtin misbehaves.
#define MLA_TRY_TDM 1

#if MLA_TRY_TDM && defined(__has_builtin)
#if __has_builtin(__builtin_amdgcn_tensor_load_to_lds)
#define HAVE_TDM 1
#endif
#endif
#ifndef HAVE_TDM
#define HAVE_TDM 0
#endif

static __device__ __forceinline__ __bf16 f32_to_bf16(float f) {
    unsigned u = __builtin_bit_cast(unsigned, f);
    u += 0x7FFFu + ((u >> 16) & 1u);             // round-to-nearest-even
    unsigned short h = (unsigned short)(u >> 16);
    return __builtin_bit_cast(__bf16, h);
}
static __device__ __forceinline__ float bf16_to_f32(__bf16 h) {
    unsigned short b = __builtin_bit_cast(unsigned short, h);
    unsigned u = ((unsigned)b) << 16;
    return __builtin_bit_cast(float, u);
}

static __device__ __forceinline__ v8f wmma_bf16(v16bf a, v16bf b, v8f c) {
    return __builtin_amdgcn_wmma_f32_16x16x32_bf16(
        false, a, false, b, (short)0, c, false, false);
}

#if HAVE_TDM
// TDM 2-D tile load: `rows` rows of `rowdw` dwords, global row stride
// `stridedw` dwords, into contiguous LDS at byte address laddr.
// D# per CDNA5 ISA:
//   group0: {count=1, lds_addr, global_addr[56:0], type=2}
//   group1: {data_size=4B, tensor_dim0=rowdw, tensor_dim1=rows,
//            tile_dim0=rowdw, tile_dim1=rows, tensor_dim0_stride=stridedw}
//   groups 2/3 zero (<= 2-D tensor).
static __device__ __forceinline__ void tdm_load_tile(
        unsigned laddr, const void* gptr, unsigned rowdw, unsigned rows,
        unsigned stridedw) {
    unsigned long long ga = (unsigned long long)(size_t)gptr;
    u32x4 g0;
    g0[0] = 1u;                                                // count=1
    g0[1] = laddr;                                             // lds_addr
    g0[2] = (unsigned)ga;                                      // global_addr[31:0]
    g0[3] = (unsigned)((ga >> 32) & 0x01FFFFFFu) | (2u << 30); // addr[56:32], type=2
    i32x8 g1;
    g1[0] = (int)(2u << 16);                                   // data_size=4B
    g1[1] = (int)((rowdw & 0xFFFFu) << 16);                    // tensor_dim0[15:0]
    g1[2] = (int)(((rowdw >> 16) & 0xFFFFu) | ((rows & 0xFFFFu) << 16)); // dim0 hi, dim1 lo
    g1[3] = (int)(((rows >> 16) & 0xFFFFu) | ((rowdw & 0xFFFFu) << 16)); // dim1 hi, tile_dim0
    g1[4] = (int)(rows & 0xFFFFu);                             // tile_dim1 (tile_dim2=0)
    g1[5] = (int)stridedw;                                     // dim0_stride[31:0]
    g1[6] = 0;                                                 // stride hi, dim1_stride lo
    g1[7] = 0;
    i32x4 gz = {0, 0, 0, 0};
#if __clang_major__ >= 23
    i32x8 gz8 = {0, 0, 0, 0, 0, 0, 0, 0};
    __builtin_amdgcn_tensor_load_to_lds(g0, g1, gz, gz, gz8, 0);
#else
    __builtin_amdgcn_tensor_load_to_lds(g0, g1, gz, gz, 0);
#endif
}
#endif

// A fragment: 16x32 bf16, row-major source (stride in elements).
static __device__ __forceinline__ v16bf load_a_frag(const __bf16* base, int stride, int lane) {
    const int m  = lane & 15;
    const int kh = (lane >> 4) & 1;
    v16bf a;
#pragma unroll
    for (int j = 0; j < 8; ++j) {
        int kb = kh * 16 + j * 2;
        a[2 * j]     = base[m * stride + kb];
        a[2 * j + 1] = base[m * stride + kb + 1];
    }
    return a;
}

// B fragment: 32x16 bf16 from row-major [K][N] tile (stride = N in elements).
static __device__ __forceinline__ v16bf load_b_frag(const __bf16* base, int stride, int lane) {
    const int n  = lane & 15;
    const int kh = (lane >> 4) & 1;
    v16bf b;
#pragma unroll
    for (int j = 0; j < 8; ++j) {
        int k = kh * 16 + 2 * j;
        b[2 * j]     = base[(k)     * stride + n];
        b[2 * j + 1] = base[(k + 1) * stride + n];
    }
    return b;
}

// B fragment where the source is row-major [N][K] (transposed use).
static __device__ __forceinline__ v16bf load_bT_frag(const __bf16* base, int stride, int lane) {
    const int n  = lane & 15;
    const int kh = (lane >> 4) & 1;
    v16bf b;
#pragma unroll
    for (int j = 0; j < 8; ++j) {
        int k = kh * 16 + 2 * j;
        b[2 * j]     = base[n * stride + k];
        b[2 * j + 1] = base[n * stride + k + 1];
    }
    return b;
}

// ---------------------------------------------------------------------------
__global__ void cvt_f32_bf16(const float* __restrict__ in, __bf16* __restrict__ out, size_t n) {
    size_t i = (size_t)blockIdx.x * blockDim.x + threadIdx.x;
    size_t s = (size_t)gridDim.x * blockDim.x;
    for (; i < n; i += s) out[i] = f32_to_bf16(in[i]);
}

// ---------------------------------------------------------------------------
// WMMA GEMM: C[M,N] = A[M,K] * B[K,N], bf16 row-major inputs.
// Block tile 128 x BN (BN = 64 or 128), 256 threads = 8 waves (4m x 2n).
// Wave tile 32 x (BN/2): NF = BN/32 n-fragments, 2*NF WMMAs per 32-K step.
// Tiles staged via TDM (wave 0 issues descriptors) or vector loads.
// ---------------------------------------------------------------------------
template <int BN, bool OUT_BF16>
__global__ __launch_bounds__(256, 1) void gemm_wmma(
        const __bf16* __restrict__ A, const __bf16* __restrict__ Bm,
        void* __restrict__ C, int M, int N, int K) {
    constexpr int NF = BN / 32;          // n-fragments per wave
    __shared__ __bf16 As[128 * 32];
    __shared__ __bf16 Bs[32 * BN];

    const int tid  = threadIdx.x;
    const int lane = tid & 31;
    const int wid  = tid >> 5;     // 0..7
    const int mw   = wid >> 1;     // 0..3
    const int nw   = wid & 1;      // 0..1
    const int bm   = blockIdx.y * 128;
    const int bn   = blockIdx.x * BN;

    v8f acc[2][NF] = {};

    for (int kt = 0; kt < K; kt += 32) {
#if HAVE_TDM
        if (wid == 0) {   // one wave issues both 2-D TDM tile loads
            // A tile: 128 rows x 16 dwords, global row stride K/2 dwords
            tdm_load_tile((unsigned)(size_t)&As[0],
                          &A[(size_t)bm * K + kt], 16u, 128u, (unsigned)(K / 2));
            // B tile: 32 rows x BN/2 dwords, global row stride N/2 dwords
            tdm_load_tile((unsigned)(size_t)&Bs[0],
                          &Bm[(size_t)kt * N + bn], (unsigned)(BN / 2), 32u,
                          (unsigned)(N / 2));
            __builtin_amdgcn_s_wait_tensorcnt(0);
        }
#else
        // A tile: 128x32 bf16 = 512 uint4; 2 per thread
#pragma unroll
        for (int i = 0; i < 2; ++i) {
            int idx = tid + i * 256;
            int row = idx >> 2;
            int c8  = (idx & 3) * 8;
            *reinterpret_cast<uint4*>(&As[row * 32 + c8]) =
                *reinterpret_cast<const uint4*>(&A[(size_t)(bm + row) * K + kt + c8]);
        }
        // B tile: 32xBN bf16 = 4*BN uint4
#pragma unroll
        for (int i = tid; i < 4 * BN; i += 256) {
            int row = i / (BN / 8);
            int c8  = (i % (BN / 8)) * 8;
            *reinterpret_cast<uint4*>(&Bs[row * BN + c8]) =
                *reinterpret_cast<const uint4*>(&Bm[(size_t)(kt + row) * N + bn + c8]);
        }
#endif
        if (kt + 32 < K) {  // global_prefetch_b8 hints for next tiles
            __builtin_prefetch(&A[(size_t)(bm + (tid >> 1)) * K + kt + 32], 0, 1);
            __builtin_prefetch(&Bm[(size_t)(kt + 32 + (tid >> 3)) * N + bn], 0, 1);
        }
        __syncthreads();

        v16bf a0 = load_a_frag(&As[(mw * 32 + 0)  * 32], 32, lane);
        v16bf a1 = load_a_frag(&As[(mw * 32 + 16) * 32], 32, lane);
#pragma unroll
        for (int j = 0; j < NF; ++j) {
            v16bf bj = load_b_frag(&Bs[nw * (BN / 2) + j * 16], BN, lane);
            acc[0][j] = wmma_bf16(a0, bj, acc[0][j]);
            acc[1][j] = wmma_bf16(a1, bj, acc[1][j]);
        }
        __syncthreads();
    }

    const int hi  = lane >> 4;
    const int nlo = lane & 15;
#pragma unroll
    for (int i = 0; i < 2; ++i)
#pragma unroll
        for (int j = 0; j < NF; ++j)
#pragma unroll
            for (int r = 0; r < 8; ++r) {
                int row = bm + mw * 32 + i * 16 + r + 8 * hi;
                int col = bn + nw * (BN / 2) + j * 16 + nlo;
                if (OUT_BF16)
                    ((__bf16*)C)[(size_t)row * N + col] = f32_to_bf16(acc[i][j][r]);
                else
                    ((float*)C)[(size_t)row * N + col] = acc[i][j][r];
            }
}

// ---------------------------------------------------------------------------
__global__ void rmsnorm_bf16(const __bf16* __restrict__ in, const float* __restrict__ w,
                             __bf16* __restrict__ out, int C, int instride) {
    const int row = blockIdx.x;
    const __bf16* xr = in + (size_t)row * instride;
    float ss = 0.f;
    for (int c = threadIdx.x; c < C; c += blockDim.x) {
        float v = bf16_to_f32(xr[c]);
        ss += v * v;
    }
#pragma unroll
    for (int off = 16; off > 0; off >>= 1) ss += __shfl_xor(ss, off, 32);
    __shared__ float red[8];
    __shared__ float s_inv;
    if ((threadIdx.x & 31) == 0) red[threadIdx.x >> 5] = ss;
    __syncthreads();
    if (threadIdx.x == 0) {
        float t = 0.f;
        for (int i = 0; i < (int)(blockDim.x >> 5); ++i) t += red[i];
        s_inv = rsqrtf(t / (float)C + 1e-6f);
    }
    __syncthreads();
    float inv = s_inv;
    for (int c = threadIdx.x; c < C; c += blockDim.x)
        out[(size_t)row * C + c] = f32_to_bf16(bf16_to_f32(xr[c]) * inv * w[c]);
}

// ---------------------------------------------------------------------------
static __device__ __forceinline__ float rope_val(float x, float other, int d, int pos) {
    int i = d & 31;
    float inv = __powf(10000.f, -(float)i / 32.f);
    float ang = (float)pos * inv;
    float s = __sinf(ang), c = __cosf(ang);
    float rot = (d < 32) ? -other : other;
    return x * c + rot * s;
}

__global__ void rope_kpe(const __bf16* __restrict__ kv, __bf16* __restrict__ kpe, int T) {
    int tok = blockIdx.x, d = threadIdx.x;             // 64 threads
    int pos = tok % T;
    const __bf16* src = kv + (size_t)tok * (KV_LORA + ROPE) + KV_LORA;
    float x = bf16_to_f32(src[d]);
    float o = bf16_to_f32(src[(d < 32) ? d + 32 : d - 32]);
    kpe[(size_t)tok * ROPE + d] = f32_to_bf16(rope_val(x, o, d, pos));
}

__global__ void build_q(const __bf16* __restrict__ q, __bf16* __restrict__ qc, int T) {
    int tok = blockIdx.x, h = blockIdx.y, d = threadIdx.x;  // 192 threads
    int b = tok / T, t = tok % T;
    const __bf16* src = q + ((size_t)tok * N_HEADS + h) * HD;
    float x = bf16_to_f32(src[d]);
    float o;
    if (d < NOPE) {
        o = x;
    } else {
        int dr = d - NOPE;
        float oth = bf16_to_f32(src[NOPE + ((dr < 32) ? dr + 32 : dr - 32)]);
        o = rope_val(x, oth, dr, t);
    }
    qc[(((size_t)(b * N_HEADS + h)) * T + t) * HD + d] = f32_to_bf16(o);
}

__global__ void build_kv(const __bf16* __restrict__ kvb, const __bf16* __restrict__ kpe,
                         __bf16* __restrict__ kc, __bf16* __restrict__ vv, int T) {
    int tok = blockIdx.x, h = blockIdx.y, d = threadIdx.x;  // 256 threads
    int b = tok / T, t = tok % T;
    size_t bh = (size_t)(b * N_HEADS + h);
    const __bf16* src = kvb + ((size_t)tok * N_HEADS + h) * (NOPE + V_DIM);
    if (d < NOPE) {
        kc[(bh * T + t) * HD + d]    = src[d];
        vv[(bh * T + t) * V_DIM + d] = src[NOPE + d];
    } else if (d < HD) {
        kc[(bh * T + t) * HD + d] = kpe[(size_t)tok * ROPE + (d - NOPE)];
    }
}

// ---------------------------------------------------------------------------
// Flash attention (causal). Grid: (T/64, H, B). Block: 128 threads = 4 waves.
// ---------------------------------------------------------------------------
__global__ __launch_bounds__(128, 1) void flash_attn(
        const __bf16* __restrict__ Q, const __bf16* __restrict__ Kc,
        const __bf16* __restrict__ V, __bf16* __restrict__ Out, int T) {
    const int qtile = blockIdx.x;
    const int h = blockIdx.y, b = blockIdx.z;
    const int tid = threadIdx.x, lane = tid & 31, wid = tid >> 5;
    const float scale = 0.07216878364870322f;   // 192^-0.5

    __shared__ __bf16 Qs[64 * HD];       // 24 KB
    __shared__ __bf16 Ks[32 * HD];       // 12 KB
    __shared__ __bf16 Vs[32 * V_DIM];    //  8 KB
    __shared__ __bf16 Ps[4][16 * 32];    //  4 KB

    const size_t bh = (size_t)(b * N_HEADS + h);
    const __bf16* Qg = Q  + (bh * T + (size_t)qtile * 64) * HD;
    const __bf16* Kg = Kc + bh * T * HD;
    const __bf16* Vg = V  + bh * T * V_DIM;

    for (int i = tid; i < 1536; i += 128)   // Q tile 64x192
        reinterpret_cast<uint4*>(Qs)[i] = reinterpret_cast<const uint4*>(Qg)[i];

    v8f O[8] = {};
    float mrow[8], lrow[8];
#pragma unroll
    for (int r = 0; r < 8; ++r) { mrow[r] = -1e30f; lrow[r] = 0.f; }

    const int hi  = lane >> 4;
    const int nlo = lane & 15;
    const int qb  = qtile * 64 + wid * 16;
    const int nkt = qtile * 2 + 2;

    for (int kt = 0; kt < nkt; ++kt) {
        __syncthreads();   // previous iteration's K/V reads done before overwrite
#if HAVE_TDM
        if (wid == 0) {    // one wave issues both TDM tile loads (contiguous: 1 row)
            tdm_load_tile((unsigned)(size_t)&Ks[0], Kg + (size_t)kt * 32 * HD,
                          (32u * HD * 2u) / 4u, 1u, (32u * HD * 2u) / 4u);
            tdm_load_tile((unsigned)(size_t)&Vs[0], Vg + (size_t)kt * 32 * V_DIM,
                          (32u * V_DIM * 2u) / 4u, 1u, (32u * V_DIM * 2u) / 4u);
            __builtin_amdgcn_s_wait_tensorcnt(0);
        }
#else
        for (int i = tid; i < 768; i += 128)   // K tile 32x192
            reinterpret_cast<uint4*>(Ks)[i] =
                reinterpret_cast<const uint4*>(Kg + (size_t)kt * 32 * HD)[i];
        for (int i = tid; i < 512; i += 128)   // V tile 32x128
            reinterpret_cast<uint4*>(Vs)[i] =
                reinterpret_cast<const uint4*>(Vg + (size_t)kt * 32 * V_DIM)[i];
#endif
        __syncthreads();

        // S = Qw (16x192) * K^T -> 16x32
        v8f s0 = {}, s1 = {};
#pragma unroll
        for (int kk = 0; kk < HD; kk += 32) {
            v16bf a  = load_a_frag(&Qs[(wid * 16) * HD + kk], HD, lane);
            v16bf b0 = load_bT_frag(&Ks[0  * HD + kk], HD, lane);
            v16bf b1 = load_bT_frag(&Ks[16 * HD + kk], HD, lane);
            s0 = wmma_bf16(a, b0, s0);
            s1 = wmma_bf16(a, b1, s1);
        }

        const int kb = kt * 32;
#pragma unroll
        for (int r = 0; r < 8; ++r) {
            int qrow = qb + r + 8 * hi;
            float v0 = s0[r] * scale, v1 = s1[r] * scale;
            if (kb + nlo      > qrow) v0 = -1e30f;   // causal
            if (kb + 16 + nlo > qrow) v1 = -1e30f;
            float mloc = fmaxf(v0, v1);
#pragma unroll
            for (int off = 1; off < 16; off <<= 1)
                mloc = fmaxf(mloc, __shfl_xor(mloc, off, 16));
            float mnew  = fmaxf(mrow[r], mloc);
            float alpha = __expf(mrow[r] - mnew);
            float p0 = __expf(v0 - mnew);
            float p1 = __expf(v1 - mnew);
            float ls = p0 + p1;
#pragma unroll
            for (int off = 1; off < 16; off <<= 1)
                ls += __shfl_xor(ls, off, 16);
            lrow[r] = lrow[r] * alpha + ls;
            mrow[r] = mnew;
#pragma unroll
            for (int n = 0; n < 8; ++n) O[n][r] *= alpha;
            Ps[wid][(r + 8 * hi) * 32 + nlo]      = f32_to_bf16(p0);
            Ps[wid][(r + 8 * hi) * 32 + 16 + nlo] = f32_to_bf16(p1);
        }
        __syncthreads();

        // O += P (16x32) * V (32x128)
        v16bf pa = load_a_frag(&Ps[wid][0], 32, lane);
#pragma unroll
        for (int n = 0; n < 8; ++n) {
            v16bf vb = load_b_frag(&Vs[n * 16], V_DIM, lane);
            O[n] = wmma_bf16(pa, vb, O[n]);
        }
    }

#pragma unroll
    for (int r = 0; r < 8; ++r) {
        float inv = 1.f / lrow[r];
        int t = qb + r + 8 * hi;
        size_t base = (((size_t)(b * T + t)) * N_HEADS + h) * V_DIM;
#pragma unroll
        for (int n = 0; n < 8; ++n)
            Out[base + n * 16 + nlo] = f32_to_bf16(O[n][r] * inv);
    }
}

// ---------------------------------------------------------------------------
extern "C" void kernel_launch(void* const* d_in, const int* in_sizes, int n_in,
                              void* d_out, int out_size, void* d_ws, size_t ws_size,
                              hipStream_t stream) {
    (void)in_sizes; (void)n_in; (void)out_size; (void)ws_size;
    const float* x     = (const float*)d_in[0];
    const float* wq_a  = (const float*)d_in[1];
    const float* q_nw  = (const float*)d_in[2];
    const float* wq_b  = (const float*)d_in[3];
    const float* wkv_a = (const float*)d_in[4];
    const float* kv_nw = (const float*)d_in[5];
    const float* wkv_b = (const float*)d_in[6];
    const float* w_out = (const float*)d_in[7];
    float* out = (float*)d_out;

    char* ws = (char*)d_ws;
    size_t off = 0;
    auto alloc_bf = [&](size_t elems) -> __bf16* {
        __bf16* p = (__bf16*)(ws + off);
        off = (off + elems * sizeof(__bf16) + 255) & ~(size_t)255;
        return p;
    };

    __bf16* xb    = alloc_bf((size_t)TOK * D_MODEL);
    __bf16* wqab  = alloc_bf((size_t)D_MODEL * Q_LORA);
    __bf16* wqbb  = alloc_bf((size_t)Q_LORA * (N_HEADS * HD));
    __bf16* wkvab = alloc_bf((size_t)D_MODEL * (KV_LORA + ROPE));
    __bf16* wkvbb = alloc_bf((size_t)KV_LORA * (N_HEADS * (NOPE + V_DIM)));
    __bf16* woutb = alloc_bf((size_t)(N_HEADS * V_DIM) * D_MODEL);
    __bf16* ql    = alloc_bf((size_t)TOK * Q_LORA);
    __bf16* kvl   = alloc_bf((size_t)TOK * (KV_LORA + ROPE));
    __bf16* qn    = alloc_bf((size_t)TOK * Q_LORA);
    __bf16* kvn   = alloc_bf((size_t)TOK * KV_LORA);
    __bf16* kpe   = alloc_bf((size_t)TOK * ROPE);
    __bf16* qf    = alloc_bf((size_t)TOK * N_HEADS * HD);
    __bf16* kvbf  = alloc_bf((size_t)TOK * N_HEADS * (NOPE + V_DIM));
    __bf16* qc    = alloc_bf((size_t)TOK * N_HEADS * HD);
    __bf16* kc    = alloc_bf((size_t)TOK * N_HEADS * HD);
    __bf16* vb    = alloc_bf((size_t)TOK * N_HEADS * V_DIM);
    __bf16* attn  = alloc_bf((size_t)TOK * N_HEADS * V_DIM);

    // 1) f32 -> bf16
    cvt_f32_bf16<<<2048, 256, 0, stream>>>(x,     xb,    (size_t)TOK * D_MODEL);
    cvt_f32_bf16<<<1024, 256, 0, stream>>>(wq_a,  wqab,  (size_t)D_MODEL * Q_LORA);
    cvt_f32_bf16<<<1024, 256, 0, stream>>>(wq_b,  wqbb,  (size_t)Q_LORA * (N_HEADS * HD));
    cvt_f32_bf16<<<512,  256, 0, stream>>>(wkv_a, wkvab, (size_t)D_MODEL * (KV_LORA + ROPE));
    cvt_f32_bf16<<<512,  256, 0, stream>>>(wkv_b, wkvbb, (size_t)KV_LORA * (N_HEADS * (NOPE + V_DIM)));
    cvt_f32_bf16<<<1024, 256, 0, stream>>>(w_out, woutb, (size_t)(N_HEADS * V_DIM) * D_MODEL);

    // 2) down-projections
    gemm_wmma<128, true><<<dim3(Q_LORA / 128, TOK / 128), 256, 0, stream>>>(
        xb, wqab, ql, TOK, Q_LORA, D_MODEL);
    gemm_wmma<64, true><<<dim3((KV_LORA + ROPE) / 64, TOK / 128), 256, 0, stream>>>(
        xb, wkvab, kvl, TOK, KV_LORA + ROPE, D_MODEL);

    // 3) norms + k_pe rope
    rmsnorm_bf16<<<TOK, 256, 0, stream>>>(ql,  q_nw,  qn,  Q_LORA, Q_LORA);
    rmsnorm_bf16<<<TOK, 256, 0, stream>>>(kvl, kv_nw, kvn, KV_LORA, KV_LORA + ROPE);
    rope_kpe<<<TOK, 64, 0, stream>>>(kvl, kpe, TT);

    // 4) up-projections
    gemm_wmma<128, true><<<dim3((N_HEADS * HD) / 128, TOK / 128), 256, 0, stream>>>(
        qn, wqbb, qf, TOK, N_HEADS * HD, Q_LORA);
    gemm_wmma<128, true><<<dim3((N_HEADS * (NOPE + V_DIM)) / 128, TOK / 128), 256, 0, stream>>>(
        kvn, wkvbb, kvbf, TOK, N_HEADS * (NOPE + V_DIM), KV_LORA);

    // 5) layout + rope for attention
    build_q <<<dim3(TOK, N_HEADS), HD,  0, stream>>>(qf, qc, TT);
    build_kv<<<dim3(TOK, N_HEADS), 256, 0, stream>>>(kvbf, kpe, kc, vb, TT);

    // 6) causal flash attention
    flash_attn<<<dim3(TT / 64, N_HEADS, BB), 128, 0, stream>>>(qc, kc, vb, attn, TT);

    // 7) output projection (f32 out)
    gemm_wmma<128, false><<<dim3(D_MODEL / 128, TOK / 128), 256, 0, stream>>>(
        attn, woutb, out, TOK, D_MODEL, N_HEADS * V_DIM);
}
